// MHAttention_20298015440938
// MI455X (gfx1250) — matrix operand order repaired and checked by reading, not compiled, not run on hardware
//
#include <hip/hip_runtime.h>

#define HD 128
#define NH 32
#define NKVH 8
#define GROUP 4

typedef __bf16 bf16;
typedef __attribute__((ext_vector_type(16))) __bf16 v16bf;
typedef __attribute__((ext_vector_type(8))) float v8f;

static __device__ inline v8f zero_v8f() {
  v8f z;
#pragma unroll
  for (int i = 0; i < 8; ++i) z[i] = 0.0f;
  return z;
}

// --- CDNA5 WMMA fragment loaders (wave32, 16x16x32 bf16) ---------------------
// A matrix 16x32 (MxK): lane L<16 holds row M=L, K in {0..15};
// lane L+16 holds row M=L, K in {16..31}; 2 bf16 per VGPR.
static __device__ inline v16bf load_frag_a(const bf16* p, int ld) {
  const int lane = threadIdx.x & 31;
  const int m = lane & 15;
  const int sel = (lane >> 4) << 4;  // 0 or 16
  v16bf a;
#pragma unroll
  for (int v = 0; v < 8; ++v) {
    int kb = 2 * v + sel;
    a[2 * v] = p[m * ld + kb];
    a[2 * v + 1] = p[m * ld + kb + 1];
  }
  return a;
}

// B matrix 32x16 (KxN) where B[k][n] = src[n][k] (transposed source, e.g. K^T / W^T):
// lanes 0-15 hold K=0..15 of column N=lane, lanes 16-31 hold K=16..31.
static __device__ inline v16bf load_frag_b_t(const bf16* p, int ld) {
  const int lane = threadIdx.x & 31;
  const int n = lane & 15;
  const int kb = (lane >> 4) << 4;
  v16bf b;
#pragma unroll
  for (int v = 0; v < 8; ++v) {
    b[2 * v] = p[n * ld + kb + 2 * v];
    b[2 * v + 1] = p[n * ld + kb + 2 * v + 1];
  }
  return b;
}

// B matrix 32x16 (KxN) where B[k][n] = src[k][n] (row-major source, e.g. V tile).
static __device__ inline v16bf load_frag_b_n(const bf16* p, int ld) {
  const int lane = threadIdx.x & 31;
  const int n = lane & 15;
  const int kb = (lane >> 4) << 4;
  v16bf b;
#pragma unroll
  for (int v = 0; v < 8; ++v) {
    b[2 * v] = p[(kb + 2 * v) * ld + n];
    b[2 * v + 1] = p[(kb + 2 * v + 1) * ld + n];
  }
  return b;
}

static __device__ inline v8f wmma_bf16(v16bf a, v16bf b, v8f c) {
  return __builtin_amdgcn_wmma_f32_16x16x32_bf16(false, a, false, b, (short)0,
                                                 c, false, false);
}

// --- async global -> LDS (gfx1250, tracked by ASYNCcnt) ----------------------
// Generic pointers to LDS carry the LDS byte offset in addr[31:0] (aperture
// rule, ISA 10.2), which is exactly what the async op's LDS-address VGPR wants.
static __device__ inline void async_copy_b128(void* lds, const bf16* gptr) {
  unsigned lo = (unsigned)(uintptr_t)lds;
  asm volatile("global_load_async_to_lds_b128 %0, %1, off"
               :
               : "v"(lo), "v"(gptr)
               : "memory");
}

// --- helper kernels ----------------------------------------------------------
__global__ void f2bf_kernel(const float* __restrict__ in, bf16* __restrict__ out,
                            size_t n) {
  size_t i = (size_t)blockIdx.x * blockDim.x + threadIdx.x;
  if (i < n) out[i] = (bf16)in[i];
}

static __device__ inline float llama3_freq(int j) {
  const float base = 500000.0f;
  float inv = powf(base, -(2.0f * (float)j) / 128.0f);
  const float factor = 32.0f, low_f = 1.0f, high_f = 4.0f, old_ctx = 8192.0f;
  const float PI2 = 6.283185307179586f;
  float wavelen = PI2 / inv;
  float low_wl = old_ctx / low_f;
  float high_wl = old_ctx / high_f;
  float inv_llama = (wavelen > low_wl) ? inv / factor : inv;
  float smooth = (old_ctx / wavelen - low_f) / (high_f - low_f);
  float smoothed = (1.0f - smooth) * inv_llama / factor + smooth * inv_llama;
  bool is_med = !(wavelen < high_wl) && !(wavelen > low_wl);
  return is_med ? smoothed : inv_llama;
}

// in: fp32 [B*T, nh*128]; out: bf16 [B, nh, T, 128] with RoPE applied (+scale)
__global__ void rope_pack_kernel(const float* __restrict__ in,
                                 bf16* __restrict__ out, int B, int T, int nh,
                                 float scale) {
  size_t idx = (size_t)blockIdx.x * blockDim.x + threadIdx.x;
  size_t total = (size_t)B * T * nh * 64;
  if (idx >= total) return;
  int j = (int)(idx & 63);
  size_t r = idx >> 6;
  int h = (int)(r % nh); r /= nh;
  int t = (int)(r % T);
  int b = (int)(r / T);
  float f = llama3_freq(j);
  float ang = (float)t * f;
  float c = cosf(ang), s = sinf(ang);
  const float* row = in + ((size_t)b * T + t) * ((size_t)nh * HD) + (size_t)h * HD;
  float x1 = row[j], x2 = row[j + 64];
  bf16* orow = out + (((size_t)(b * nh + h) * T) + t) * HD;
  orow[j] = (bf16)((x1 * c - x2 * s) * scale);
  orow[j + 64] = (bf16)((x2 * c + x1 * s) * scale);
}

// in: fp32 [B*T, NKVH*128]; out: bf16 [B, NKVH, T, 128]
__global__ void pack_v_kernel(const float* __restrict__ in,
                              bf16* __restrict__ out, int B, int T) {
  size_t idx = (size_t)blockIdx.x * blockDim.x + threadIdx.x;
  size_t total = (size_t)B * T * NKVH * HD;
  if (idx >= total) return;
  int d = (int)(idx & 127);
  size_t r = idx >> 7;
  int kh = (int)(r % NKVH); r /= NKVH;
  int t = (int)(r % T);
  int b = (int)(r / T);
  out[((size_t)(b * NKVH + kh) * T + t) * HD + d] =
      (bf16)in[((size_t)b * T + t) * ((size_t)NKVH * HD) + (size_t)kh * HD + d];
}

// --- GEMM: C[M,N] = A[M,K] (bf16) x W[N,K]^T (bf16), fp32 accumulate --------
// Double-buffered LDS staged via global_load_async_to_lds_b128 (ASYNCcnt).
__global__ __launch_bounds__(256) void gemm_bf16_nt(
    const bf16* __restrict__ A, const bf16* __restrict__ W,
    float* __restrict__ C, int M, int N, int K) {
  __shared__ bf16 As[2][128][40];
  __shared__ bf16 Ws[2][128][40];
  const int m0 = blockIdx.y * 128;
  const int n0 = blockIdx.x * 128;
  const int tid = threadIdx.x;
  const int wave = tid >> 5;
  const int wm = wave >> 1;  // 0..3 -> 32-row slice
  const int wn = wave & 1;   // 0..1 -> 64-col slice

  const int r_ld = tid >> 2;        // 0..63? no: tid/4 -> 0..63 for i=0, +64 for i=1
  const int s_ld = tid & 3;

  v8f acc[2][4];
#pragma unroll
  for (int i = 0; i < 2; ++i)
#pragma unroll
    for (int j = 0; j < 4; ++j) acc[i][j] = zero_v8f();

  // issue async fill of buffer `buf` with the K-slice starting at k0
  auto fill = [&](int buf, int k0) {
#pragma unroll
    for (int i = 0; i < 2; ++i) {
      int r = r_ld + i * 64;
      async_copy_b128(&As[buf][r][s_ld * 8],
                      A + (size_t)(m0 + r) * K + k0 + s_ld * 8);
      async_copy_b128(&Ws[buf][r][s_ld * 8],
                      W + (size_t)(n0 + r) * K + k0 + s_ld * 8);
    }
  };

  fill(0, 0);
  const int NIT = K / 32;
  for (int it = 0; it < NIT; ++it) {
    const int cur = it & 1;
    if (it + 1 < NIT) {
      fill(cur ^ 1, (it + 1) * 32);
      // 8 async loads in flight; in-order completion => <=4 means cur is done
      asm volatile("s_wait_asynccnt 0x4" ::: "memory");
    } else {
      asm volatile("s_wait_asynccnt 0x0" ::: "memory");
    }
    __syncthreads();  // cur buffer visible to all waves

    v16bf a[2], b[4];
#pragma unroll
    for (int i = 0; i < 2; ++i)
      a[i] = load_frag_a(&As[cur][wm * 32 + i * 16][0], 40);
#pragma unroll
    for (int j = 0; j < 4; ++j)
      b[j] = load_frag_b_t(&Ws[cur][wn * 64 + j * 16][0], 40);
#pragma unroll
    for (int i = 0; i < 2; ++i)
#pragma unroll
      for (int j = 0; j < 4; ++j) acc[i][j] = wmma_bf16(a[i], b[j], acc[i][j]);

    __syncthreads();  // all waves done reading cur before it is refilled
  }

  const int lane = tid & 31;
  const int cl = lane & 15;
  const int rh = (lane >> 4) * 8;
#pragma unroll
  for (int i = 0; i < 2; ++i)
#pragma unroll
    for (int j = 0; j < 4; ++j) {
      int row = m0 + wm * 32 + i * 16 + rh;
      int col = n0 + wn * 64 + j * 16 + cl;
      float* cp = C + (size_t)row * N + col;
#pragma unroll
      for (int v = 0; v < 8; ++v) cp[(size_t)v * N] = acc[i][j][v];
    }
}

// --- Flash attention (causal, GQA). Q:[B,H,T,128] K/V:[B,KVH,T,128] bf16 ----
// Out: bf16 [B*T, H*128]
__global__ __launch_bounds__(128) void attn_fwd(
    const bf16* __restrict__ Q, const bf16* __restrict__ Kt,
    const bf16* __restrict__ Vt, bf16* __restrict__ O, int T) {
  __shared__ bf16 Qs[64][136];
  __shared__ bf16 Ks[64][136];
  __shared__ bf16 Vs[64][136];
  __shared__ bf16 Ps[4][16][64];

  const int tid = threadIdx.x;
  const int wave = tid >> 5;
  const int lane = tid & 31;
  const int bh = blockIdx.y;
  const int b = bh / NH;
  const int h = bh % NH;
  const int kvh = h / GROUP;
  const int q0 = blockIdx.x * 64;

  const bf16* Qg = Q + ((size_t)(b * NH + h) * T + q0) * HD;
  const bf16* Kg = Kt + ((size_t)(b * NKVH + kvh) * T) * HD;
  const bf16* Vg = Vt + ((size_t)(b * NKVH + kvh) * T) * HD;

#pragma unroll
  for (int i = 0; i < 8; ++i) {
    int idx = i * 128 + tid;
    int r = idx >> 4, s = idx & 15;
    *(uint4*)&Qs[r][s * 8] = *(const uint4*)(Qg + (size_t)r * HD + s * 8);
  }
  __syncthreads();

  v16bf qa[4];
#pragma unroll
  for (int kk = 0; kk < 4; ++kk)
    qa[kk] = load_frag_a(&Qs[wave * 16][kk * 32], 136);

  v8f oacc[8];
#pragma unroll
  for (int i = 0; i < 8; ++i) oacc[i] = zero_v8f();
  float mrow[8], lrow[8];
#pragma unroll
  for (int v = 0; v < 8; ++v) { mrow[v] = -1e30f; lrow[v] = 0.0f; }

  const int cl = lane & 15;
  const int rh = (lane >> 4) * 8;
  const int kb_max = blockIdx.x;

  for (int kb = 0; kb <= kb_max; ++kb) {
    __syncthreads();
#pragma unroll
    for (int i = 0; i < 8; ++i) {
      int idx = i * 128 + tid;
      int r = idx >> 4, s = idx & 15;
      *(uint4*)&Ks[r][s * 8] =
          *(const uint4*)(Kg + (size_t)(kb * 64 + r) * HD + s * 8);
      *(uint4*)&Vs[r][s * 8] =
          *(const uint4*)(Vg + (size_t)(kb * 64 + r) * HD + s * 8);
    }
    __syncthreads();

    // S = Q K^T : 16(q) x 64(k) per wave
    v8f sfr[4];
#pragma unroll
    for (int nt = 0; nt < 4; ++nt) {
      v8f s = zero_v8f();
#pragma unroll
      for (int kk = 0; kk < 4; ++kk) {
        v16bf bk = load_frag_b_t(&Ks[nt * 16][kk * 32], 136);
        s = wmma_bf16(qa[kk], bk, s);
      }
      sfr[nt] = s;
    }

    if (kb == kb_max) {  // diagonal block: causal mask
#pragma unroll
      for (int nt = 0; nt < 4; ++nt)
#pragma unroll
        for (int v = 0; v < 8; ++v) {
          int qg = q0 + wave * 16 + rh + v;
          int kg = kb * 64 + nt * 16 + cl;
          if (kg > qg) sfr[nt][v] = -1e30f;
        }
    }

    // online softmax (rows live in one VGPR across a 16-lane group)
#pragma unroll
    for (int v = 0; v < 8; ++v) {
      float mx = sfr[0][v];
#pragma unroll
      for (int nt = 1; nt < 4; ++nt) mx = fmaxf(mx, sfr[nt][v]);
#pragma unroll
      for (int off = 1; off < 16; off <<= 1)
        mx = fmaxf(mx, __shfl_xor(mx, off, 32));
      float mnew = fmaxf(mrow[v], mx);
      float corr = __expf(mrow[v] - mnew);
      mrow[v] = mnew;
      float rs = 0.0f;
#pragma unroll
      for (int nt = 0; nt < 4; ++nt) {
        float p = __expf(sfr[nt][v] - mnew);
        sfr[nt][v] = p;
        rs += p;
      }
#pragma unroll
      for (int off = 1; off < 16; off <<= 1) rs += __shfl_xor(rs, off, 32);
      lrow[v] = lrow[v] * corr + rs;
#pragma unroll
      for (int nt = 0; nt < 8; ++nt) oacc[nt][v] *= corr;
    }

    // restripe P (C-layout -> A-layout) through per-wave LDS scratch
#pragma unroll
    for (int nt = 0; nt < 4; ++nt)
#pragma unroll
      for (int v = 0; v < 8; ++v)
        Ps[wave][rh + v][nt * 16 + cl] = (bf16)sfr[nt][v];
    asm volatile("s_wait_dscnt 0" ::: "memory");

    // O += P * V
#pragma unroll
    for (int kk = 0; kk < 2; ++kk) {
      v16bf pa = load_frag_a(&Ps[wave][0][kk * 32], 64);
#pragma unroll
      for (int nt = 0; nt < 8; ++nt) {
        v16bf vb = load_frag_b_n(&Vs[kk * 32][nt * 16], 136);
        oacc[nt] = wmma_bf16(pa, vb, oacc[nt]);
      }
    }
  }

  // normalize and write to [B*T, NH*HD]
#pragma unroll
  for (int v = 0; v < 8; ++v) {
    float inv = 1.0f / lrow[v];
    int qrow = q0 + wave * 16 + rh + v;
    bf16* orow = O + ((size_t)(b * T + qrow)) * ((size_t)NH * HD) + (size_t)h * HD;
#pragma unroll
    for (int nt = 0; nt < 8; ++nt)
      orow[nt * 16 + cl] = (bf16)(oacc[nt][v] * inv);
  }
}

// --- host launcher -----------------------------------------------------------
extern "C" void kernel_launch(void* const* d_in, const int* in_sizes, int n_in,
                              void* d_out, int out_size, void* d_ws,
                              size_t ws_size, hipStream_t stream) {
  (void)in_sizes; (void)n_in; (void)out_size; (void)ws_size;
  const float* x = (const float*)d_in[0];
  const float* Wq = (const float*)d_in[1];
  const float* Wk = (const float*)d_in[2];
  const float* Wv = (const float*)d_in[3];
  const float* Wo = (const float*)d_in[4];
  float* out = (float*)d_out;

  const int B = 2, T = 2048, D = 4096, Dkv = NKVH * HD;  // 1024
  const int M = B * T;                                   // 4096

  char* ws = (char*)d_ws;
  size_t off = 0;
  auto alloc = [&](size_t bytes) -> void* {
    void* p = ws + off;
    off += (bytes + 255) & ~(size_t)255;
    return p;
  };
  bf16* xb = (bf16*)alloc((size_t)M * D * 2);
  bf16* Wqb = (bf16*)alloc((size_t)D * D * 2);
  bf16* Wkb = (bf16*)alloc((size_t)Dkv * D * 2);
  bf16* Wvb = (bf16*)alloc((size_t)Dkv * D * 2);
  bf16* Wob = (bf16*)alloc((size_t)D * D * 2);
  float* Qf = (float*)alloc((size_t)M * D * 4);
  float* Kf = (float*)alloc((size_t)M * Dkv * 4);
  float* Vf = (float*)alloc((size_t)M * Dkv * 4);
  bf16* Qr = (bf16*)alloc((size_t)M * D * 2);
  bf16* Kr = (bf16*)alloc((size_t)M * Dkv * 2);
  bf16* Vr = (bf16*)alloc((size_t)M * Dkv * 2);
  bf16* Ab = (bf16*)alloc((size_t)M * D * 2);

  auto grid1 = [](size_t n) { return dim3((unsigned)((n + 255) / 256)); };

  { size_t n = (size_t)M * D;   f2bf_kernel<<<grid1(n), 256, 0, stream>>>(x, xb, n); }
  { size_t n = (size_t)D * D;   f2bf_kernel<<<grid1(n), 256, 0, stream>>>(Wq, Wqb, n); }
  { size_t n = (size_t)Dkv * D; f2bf_kernel<<<grid1(n), 256, 0, stream>>>(Wk, Wkb, n); }
  { size_t n = (size_t)Dkv * D; f2bf_kernel<<<grid1(n), 256, 0, stream>>>(Wv, Wvb, n); }
  { size_t n = (size_t)D * D;   f2bf_kernel<<<grid1(n), 256, 0, stream>>>(Wo, Wob, n); }

  gemm_bf16_nt<<<dim3(D / 128, M / 128), 256, 0, stream>>>(xb, Wqb, Qf, M, D, D);
  gemm_bf16_nt<<<dim3(Dkv / 128, M / 128), 256, 0, stream>>>(xb, Wkb, Kf, M, Dkv, D);
  gemm_bf16_nt<<<dim3(Dkv / 128, M / 128), 256, 0, stream>>>(xb, Wvb, Vf, M, Dkv, D);

  const float qscale = 0.08838834764831845f;  // 1/sqrt(128)
  { size_t n = (size_t)B * T * NH * 64;
    rope_pack_kernel<<<grid1(n), 256, 0, stream>>>(Qf, Qr, B, T, NH, qscale); }
  { size_t n = (size_t)B * T * NKVH * 64;
    rope_pack_kernel<<<grid1(n), 256, 0, stream>>>(Kf, Kr, B, T, NKVH, 1.0f); }
  { size_t n = (size_t)B * T * NKVH * HD;
    pack_v_kernel<<<grid1(n), 256, 0, stream>>>(Vf, Vr, B, T); }

  attn_fwd<<<dim3(T / 64, B * NH), 128, 0, stream>>>(Qr, Kr, Vr, Ab, T);

  gemm_bf16_nt<<<dim3(D / 128, M / 128), 256, 0, stream>>>(Ab, Wob, out, M, D, D);
}
